// CutCrossEntropyLoss_58334245814932
// MI455X (gfx1250) — matrix operand order repaired and checked by reading, not compiled, not run on hardware
//
#include <hip/hip_runtime.h>
#include <hip/hip_bf16.h>
#include <math.h>

// Problem constants (from reference)
#define BB 2
#define SS 4096
#define DD 2048
#define VV 50257
#define NN (BB * (SS - 1))   // 8190 rows
#define IGNORE_IDX (-100)

#define MT 32                // rows per workgroup (2 x 16-row WMMA sub-tiles)
#define NPAD 8192            // padded row count for workspace partials
#define NCH ((VV + 127) / 128)        // 393 vocab chunks of 128
#define CHSPLIT ((NCH + 1) / 2)       // 197 chunks per vocab split

typedef __attribute__((ext_vector_type(16))) __bf16 v16bf;
typedef __attribute__((ext_vector_type(8)))  __bf16 v8bf;
typedef __attribute__((ext_vector_type(4)))  __bf16 v4bf;
typedef __attribute__((ext_vector_type(8)))  float  v8f;

__device__ __forceinline__ void online_merge(float& m, float& s, float mo, float so) {
    float nm = fmaxf(m, mo);
    if (nm == -INFINITY) { m = nm; s = 0.0f; return; }   // both empty
    s = s * __expf(m - nm) + so * __expf(mo - nm);
    m = nm;
}

__global__ void cce_init_ws(float* ws) {
    if (threadIdx.x < 2) ws[threadIdx.x] = 0.0f;
}

// GEMM + per-vocab-split online logsumexp partials. grid = (row tiles, 2 vocab splits)
__global__ __launch_bounds__(256) void cce_fused_kernel(
    const float* __restrict__ emb,     // [B, S, D] fp32
    const float* __restrict__ weight,  // [V, D]    fp32
    const float* __restrict__ bias,    // [V]       fp32
    const int*   __restrict__ labels,  // [B, S]
    float* __restrict__ Pm,            // [2][NPAD] partial running max
    float* __restrict__ Ps,            // [2][NPAD] partial running sum-exp
    float* __restrict__ Pt)            // [2][NPAD] partial target logit
{
    __shared__ __attribute__((aligned(16))) __bf16 sA[MT * DD];  // 128 KB bf16 A tile

    const int tid    = threadIdx.x;
    const int lane   = tid & 31;
    const int waveId = tid >> 5;          // 0..7
    const int mbase  = blockIdx.x * MT;   // first global row of this tile
    const int vsplit = blockIdx.y;        // 0/1: which half of the vocab

    // ---- Stage A tile: MT rows x D, fp32 -> bf16 into LDS (coalesced) ----
    for (int idx = tid; idx < MT * (DD / 4); idx += 256) {
        int row = idx / (DD / 4);
        int c4  = idx - row * (DD / 4);
        int g   = mbase + row;
        float4 v = make_float4(0.0f, 0.0f, 0.0f, 0.0f);
        if (g < NN) {
            int b  = g / (SS - 1);
            int sp = g - b * (SS - 1);
            v = *(const float4*)(emb + ((size_t)(b * SS + sp)) * DD + (size_t)c4 * 4);
        }
        v4bf h;
        h[0] = (__bf16)v.x; h[1] = (__bf16)v.y; h[2] = (__bf16)v.z; h[3] = (__bf16)v.w;
        *(v4bf*)(&sA[row * DD + c4 * 4]) = h;   // ds_store_b64
    }
    __syncthreads();

    const int arow  = lane & 15;              // A fragment: M row index (within sub-tile)
    const int khalf = (lane >> 4) & 1;        // lane half selects K sub-block
    const int koffA = khalf * 16;             // A: K run  {koffA .. koffA+15}
    const int koffB = khalf * 16;             // B: K run  {koffB .. koffB+15}
    const int bcol  = lane & 15;              // B/C column within wave tile
    const int rbase = khalf ? 8 : 0;          // C row base within sub-tile

    // Per-lane labels for its 8 output rows in each sub-tile
    int lab[2][8];
    #pragma unroll
    for (int st = 0; st < 2; ++st) {
        #pragma unroll
        for (int r = 0; r < 8; ++r) {
            int g  = mbase + st * 16 + rbase + r;
            int lv = -2;
            if (g < NN) {
                int b  = g / (SS - 1);
                int sp = g - b * (SS - 1);
                lv = labels[b * SS + sp + 1];     // shifted labels
            }
            lab[st][r] = lv;
        }
    }

    float m8[2][8], s8[2][8], t8[2][8];
    #pragma unroll
    for (int st = 0; st < 2; ++st)
        #pragma unroll
        for (int r = 0; r < 8; ++r) { m8[st][r] = -INFINITY; s8[st][r] = 0.0f; t8[st][r] = 0.0f; }

    const int ch0   = vsplit * CHSPLIT;
    const int chEnd = min(NCH, ch0 + CHSPLIT);
    for (int ch = ch0; ch < chEnd; ++ch) {
        const int  col      = ch * 128 + (waveId << 4) + bcol;  // global vocab col
        const bool colValid = (col < VV);
        const int  colC     = colValid ? col : (VV - 1);        // clamp: loads always legal
        const float* wrow   = weight + (size_t)colC * DD;

        if (colC + 128 < VV)                                    // warm next chunk
            __builtin_prefetch(wrow + (size_t)128 * DD, 0, 1);  // global_prefetch_b8

        v8f acc[2];
        #pragma unroll
        for (int st = 0; st < 2; ++st)
            acc[st] = (v8f){0.f, 0.f, 0.f, 0.f, 0.f, 0.f, 0.f, 0.f};

        for (int kb = 0; kb < DD; kb += 32) {
            // --- B fragment from global fp32 -> bf16 (16 contiguous K), branch-free ---
            const float* wp = wrow + kb + koffB;
            float4 f0 = *(const float4*)(wp + 0);
            float4 f1 = *(const float4*)(wp + 4);
            float4 f2 = *(const float4*)(wp + 8);
            float4 f3 = *(const float4*)(wp + 12);
            v16bf b;
            b[0]  = (__bf16)f0.x; b[1]  = (__bf16)f0.y; b[2]  = (__bf16)f0.z; b[3]  = (__bf16)f0.w;
            b[4]  = (__bf16)f1.x; b[5]  = (__bf16)f1.y; b[6]  = (__bf16)f1.z; b[7]  = (__bf16)f1.w;
            b[8]  = (__bf16)f2.x; b[9]  = (__bf16)f2.y; b[10] = (__bf16)f2.z; b[11] = (__bf16)f2.w;
            b[12] = (__bf16)f3.x; b[13] = (__bf16)f3.y; b[14] = (__bf16)f3.z; b[15] = (__bf16)f3.w;

            // --- 2 sub-tiles of A reuse the same B fragment ---
            #pragma unroll
            for (int st = 0; st < 2; ++st) {
                const __bf16* ap = &sA[(st * 16 + arow) * DD + kb + koffA];
                v8bf alo = *(const v8bf*)ap;          // K = kb+koffA    .. +7
                v8bf ahi = *(const v8bf*)(ap + 8);    // K = kb+8+koffA  .. +7
                v16bf a;
                #pragma unroll
                for (int i = 0; i < 8; ++i) { a[i] = alo[i]; a[8 + i] = ahi[i]; }

                acc[st] = __builtin_amdgcn_wmma_f32_16x16x32_bf16(
                    false, a, false, b, (short)0, acc[st], false, false);
            }
        }

        // --- bias + online logsumexp + target capture (tail cols excluded here) ---
        if (colValid) {
            float bv = bias[col];
            #pragma unroll
            for (int st = 0; st < 2; ++st) {
                #pragma unroll
                for (int r = 0; r < 8; ++r) {
                    float x  = acc[st][r] + bv;
                    float nm = fmaxf(m8[st][r], x);
                    s8[st][r] = s8[st][r] * __expf(m8[st][r] - nm) + __expf(x - nm);
                    m8[st][r] = nm;
                    if (col == lab[st][r]) t8[st][r] += x;
                }
            }
        }
    }

    // ---- intra-wave reduce across the 16 columns (stays within lane-half) ----
    #pragma unroll
    for (int st = 0; st < 2; ++st) {
        #pragma unroll
        for (int r = 0; r < 8; ++r) {
            #pragma unroll
            for (int off = 1; off < 16; off <<= 1) {
                float mo = __shfl_xor(m8[st][r], off, 32);
                float so = __shfl_xor(s8[st][r], off, 32);
                online_merge(m8[st][r], s8[st][r], mo, so);
                t8[st][r] += __shfl_xor(t8[st][r], off, 32);
            }
        }
    }

    // ---- cross-wave merge through LDS (A buffer is dead; reuse it) ----
    __syncthreads();
    float* mM = (float*)sA;        // [8 waves][MT rows]
    float* sS = mM + 8 * MT;
    float* tT = mM + 16 * MT;
    if (bcol == 0) {
        #pragma unroll
        for (int st = 0; st < 2; ++st) {
            #pragma unroll
            for (int r = 0; r < 8; ++r) {
                int row = st * 16 + rbase + r;
                mM[waveId * MT + row] = m8[st][r];
                sS[waveId * MT + row] = s8[st][r];
                tT[waveId * MT + row] = t8[st][r];
            }
        }
    }
    __syncthreads();

    if (tid < MT) {                // first wave: one lane per row; plain stores, no atomics
        float m = -INFINITY, s = 0.0f, t = 0.0f;
        #pragma unroll
        for (int w = 0; w < 8; ++w) {
            online_merge(m, s, mM[w * MT + tid], sS[w * MT + tid]);
            t += tT[w * MT + tid];
        }
        int g = mbase + tid;                 // < NPAD by construction
        Pm[vsplit * NPAD + g] = m;
        Ps[vsplit * NPAD + g] = s;
        Pt[vsplit * NPAD + g] = t;
    }
}

// Merge the two vocab-split partials per row, mask, and accumulate sum/count.
__global__ __launch_bounds__(256) void cce_reduce(
    const float* __restrict__ Pm, const float* __restrict__ Ps,
    const float* __restrict__ Pt, const int* __restrict__ labels,
    float* __restrict__ ws_acc)
{
    int   g   = blockIdx.x * 256 + threadIdx.x;
    float nll = 0.0f, cnt = 0.0f;
    if (g < NN) {
        float m = -INFINITY, s = 0.0f, t = 0.0f;
        #pragma unroll
        for (int sp = 0; sp < 2; ++sp) {
            online_merge(m, s, Pm[sp * NPAD + g], Ps[sp * NPAD + g]);
            t += Pt[sp * NPAD + g];
        }
        int b  = g / (SS - 1);
        int p  = g - b * (SS - 1);
        int lv = labels[b * SS + p + 1];
        if (lv != IGNORE_IDX) {
            nll = (m + __logf(s)) - t;
            cnt = 1.0f;
        }
    }
    #pragma unroll
    for (int off = 16; off >= 1; off >>= 1) {
        nll += __shfl_xor(nll, off, 32);
        cnt += __shfl_xor(cnt, off, 32);
    }
    if ((threadIdx.x & 31) == 0) {
        atomicAdd(&ws_acc[0], nll);
        atomicAdd(&ws_acc[1], cnt);
    }
}

__global__ void cce_finalize(const float* __restrict__ ws_acc, float* __restrict__ out) {
    if (threadIdx.x == 0 && blockIdx.x == 0)
        out[0] = ws_acc[0] / fmaxf(ws_acc[1], 1.0f);
}

extern "C" void kernel_launch(void* const* d_in, const int* in_sizes, int n_in,
                              void* d_out, int out_size, void* d_ws, size_t ws_size,
                              hipStream_t stream) {
    const float* emb    = (const float*)d_in[0];  // [B,S,D] fp32
    const float* weight = (const float*)d_in[1];  // [V,D]   fp32
    const float* bias   = (const float*)d_in[2];  // [V]     fp32
    const int*   labels = (const int*)d_in[3];    // [B,S]   int
    float*       out    = (float*)d_out;          // scalar

    // Workspace layout (floats): [0..1] sum/count accumulators, then
    // 3 partial arrays of [2][NPAD]  (total ~192.2 KB; ws is preallocated scratch)
    float* ws  = (float*)d_ws;
    float* acc = ws;
    float* Pm  = ws + 8;
    float* Ps  = Pm + 2 * NPAD;
    float* Pt  = Ps + 2 * NPAD;

    cce_init_ws<<<1, 32, 0, stream>>>(acc);

    dim3 grid((NN + MT - 1) / MT, 2);             // 256 row tiles x 2 vocab splits = 512 WGs
    cce_fused_kernel<<<grid, 256, 0, stream>>>(emb, weight, bias, labels, Pm, Ps, Pt);

    cce_reduce<<<(NN + 255) / 256, 256, 0, stream>>>(Pm, Ps, Pt, labels, acc);

    cce_finalize<<<1, 32, 0, stream>>>(acc, out);
}
